// CAGroup3DHead_63118839382060
// MI455X (gfx1250) — matrix operand order repaired and checked by reading, not compiled, not run on hardware
//
#include <hip/hip_runtime.h>

typedef __attribute__((ext_vector_type(16))) __bf16 v16bf;
typedef __attribute__((ext_vector_type(8)))  float  v8f;

#define N_VOX 100000
#define C 128
#define NCLS 18
#define NREG 6
#define VSZ 0.04f
#define NOUT 151
#define HS (1 << 18)
#define HMASK (HS - 1)
#define HASHD 260
// logit(0.15) = log(0.15/0.85)
#define SEM_LOGIT (-1.7346011f)

__device__ __forceinline__ int hash_slot(int key) {
  return (int)(((unsigned)key * 2654435761u) >> 14);   // 18-bit slot
}

// ---------------- init: hash table + coord min/max ----------------
__global__ void init_kernel(int* hKey, int* hVal, int* mm) {
  int i = blockIdx.x * blockDim.x + threadIdx.x;
  if (i < HS) { hKey[i] = -1; hVal[i] = 0x7fffffff; }
  if (i < 3)  { mm[i] = 0x7fffffff; mm[3 + i] = (int)0x80000000; }
}

// ---------------- weight swizzle to bf16 B-fragment layout ----------------
// dst element index encodes (m, kc, nt, lane, h):
//   m  : matrix (0..26 = fo_w taps, 27 = off_w1, 28 = off_w2)
//   kc : K-chunk (32 rows of c_in per chunk)
//   nt : 16-col output tile (wave id)
//   B layout: c_in = kc*32 + (lane>=16 ? 8 : 0) + (h<8 ? h : h+8) ;  c_out = nt*16 + (lane&15)
__global__ void swizzle_kernel(const float* __restrict__ fo_w,
                               const float* __restrict__ w1,
                               const float* __restrict__ w2,
                               __bf16* __restrict__ dst) {
  int idx = blockIdx.x * blockDim.x + threadIdx.x;
  if (idx >= 29 * 16384) return;
  int m    = idx >> 14;
  int r    = idx & 16383;
  int kc   = r >> 12;
  int nt   = (r >> 9) & 7;
  int lane = (r >> 4) & 31;
  int h    = r & 15;
  int cin  = kc * 32 + ((lane & 16) ? 8 : 0) + (h < 8 ? h : h + 8);
  int cout = nt * 16 + (lane & 15);
  const float* src = (m < 27) ? (fo_w + m * 16384) : ((m == 27) ? w1 : w2);
  dst[idx] = (__bf16)src[cin * C + cout];
}

// ---------------- build: hash insert + coord bounds ----------------
__global__ void build_kernel(const int* __restrict__ coords, const int* __restrict__ bidx,
                             int* hKey, int* hVal, int* mm) {
  int i = blockIdx.x * blockDim.x + threadIdx.x;
  if (i >= N_VOX) return;
  int x = coords[i * 3 + 0], y = coords[i * 3 + 1], z = coords[i * 3 + 2];
  int b = bidx[i];
  atomicMin(&mm[0], x); atomicMax(&mm[3], x);
  atomicMin(&mm[1], y); atomicMax(&mm[4], y);
  atomicMin(&mm[2], z); atomicMax(&mm[5], z);
  int key = ((b * HASHD + x + 1) * HASHD + y + 1) * HASHD + z + 1;
  int slot = hash_slot(key);
  while (true) {
    int old = atomicCAS(&hKey[slot], -1, key);
    if (old == -1 || old == key) { atomicMin(&hVal[slot], i); break; }  // lowest index wins (deterministic)
    slot = (slot + 1) & HMASK;
  }
}

// ---------------- fragment loaders ----------------
// A (16x32 bf16), ISA layout: row = lane&15, kbase = lane<16 ? 0 : 8,
// halves 0..7 -> K = kbase+h, halves 8..15 -> K = kbase+16+(h-8)
__device__ __forceinline__ v16bf load_a_frag(const __bf16* base /*16B aligned*/) {
  union { v16bf v; uint4 u[2]; } t;
  const uint4* q = (const uint4*)base;
  t.u[0] = q[0];     // +0  bytes : halves 0..7
  t.u[1] = q[2];     // +32 bytes : halves 16..23
  return t.v;
}
__device__ __forceinline__ v16bf load_b_frag(const __bf16* p /*32B aligned*/) {
  union { v16bf v; uint4 u[2]; } t;
  const uint4* q = (const uint4*)p;
  t.u[0] = q[0]; t.u[1] = q[1];
  return t.v;
}

// gather one 16x128 neighbor tile (f32 global -> bf16 LDS), 8 elems/thread
__device__ __forceinline__ void gather_tap(const float* __restrict__ feats,
                                           const int (*sNbr)[16], int k, int tid,
                                           __bf16* __restrict__ buf) {
  int idx = tid * 8;
  int r = idx >> 7, c = idx & 127;
  int n = sNbr[k][r];
  __bf16* d = &buf[r * C + c];
  if (n >= 0) {
    const float4* p = (const float4*)&feats[(size_t)n * C + c];
    float4 u0 = p[0], u1 = p[1];
    d[0] = (__bf16)u0.x; d[1] = (__bf16)u0.y; d[2] = (__bf16)u0.z; d[3] = (__bf16)u0.w;
    d[4] = (__bf16)u1.x; d[5] = (__bf16)u1.y; d[6] = (__bf16)u1.z; d[7] = (__bf16)u1.w;
  } else {
    #pragma unroll
    for (int q = 0; q < 8; ++q) d[q] = (__bf16)0.0f;
  }
}

// ---------------- fused head kernel: one block = 16 voxels ----------------
__global__ __launch_bounds__(256) void fused_kernel(
    const float* __restrict__ feats, const int* __restrict__ coords, const int* __restrict__ bidx,
    const float* __restrict__ off_g1, const float* __restrict__ off_b1,
    const float* __restrict__ off_g2, const float* __restrict__ off_b2,
    const float* __restrict__ off_w3,
    const float* __restrict__ fo_g, const float* __restrict__ fo_b,
    const float* __restrict__ sem_w, const float* __restrict__ sem_b,
    const float* __restrict__ cen_w, const float* __restrict__ cls_w,
    const float* __restrict__ cls_b, const float* __restrict__ reg_w,
    const float* __restrict__ scales,
    const int* __restrict__ hKey, const int* __restrict__ hVal, const int* __restrict__ mm,
    const __bf16* __restrict__ Wsw, float* __restrict__ out) {
  __shared__ int sC[16][4];
  __shared__ int sNbr[27][16];
  __shared__ int sHas[27];
  __shared__ int sAct[28];                     // [0..26] compacted tap list, [27] count
  __shared__ alignas(16) float  sF[16 * C];    // feats f32 (later reused for h2)
  __shared__ alignas(16) __bf16 sA[16 * C];    // feats bf16 (A for MLP1)
  __shared__ alignas(16) __bf16 sG0[16 * C];   // gather ping
  __shared__ alignas(16) __bf16 sG1[16 * C];   // gather pong
  __shared__ alignas(16) __bf16 sH[16 * C];    // h1 bf16
  __shared__ alignas(16) float  sOF[16 * C];   // off_feat f32
  __shared__ float sMask[16][NCLS];
  __shared__ float sReg[16][NREG];

  const int tid  = threadIdx.x;
  const int wave = tid >> 5;
  const int lane = tid & 31;
  const int r0   = blockIdx.x * 16;

  // ---- Stage A: coords + hasNbr init ----
  if (tid < 27) sHas[tid] = 0;
  if (tid < 16) {
    int row = r0 + tid; if (row >= N_VOX) row = N_VOX - 1;
    sC[tid][0] = coords[row * 3 + 0];
    sC[tid][1] = coords[row * 3 + 1];
    sC[tid][2] = coords[row * 3 + 2];
    sC[tid][3] = bidx[row];
  }
  __syncthreads();

  // ---- Stage B: own feats tile (f32 + bf16) ----
  {
    int idx = tid * 8;
    int r = idx >> 7, c = idx & 127;
    int row = r0 + r; if (row >= N_VOX) row = N_VOX - 1;
    const float4* p = (const float4*)&feats[(size_t)row * C + c];
    float4 u0 = p[0], u1 = p[1];
    float* f = &sF[r * C + c];
    f[0] = u0.x; f[1] = u0.y; f[2] = u0.z; f[3] = u0.w;
    f[4] = u1.x; f[5] = u1.y; f[6] = u1.z; f[7] = u1.w;
    __bf16* a = &sA[r * C + c];
    a[0] = (__bf16)u0.x; a[1] = (__bf16)u0.y; a[2] = (__bf16)u0.z; a[3] = (__bf16)u0.w;
    a[4] = (__bf16)u1.x; a[5] = (__bf16)u1.y; a[6] = (__bf16)u1.z; a[7] = (__bf16)u1.w;
  }
  // ---- Stage B': 27x16 neighbor lookups via hash ----
  for (int t = tid; t < 27 * 16; t += 256) {
    int k = t >> 4, r = t & 15;
    int dx = k / 9 - 1, dy = (k / 3) % 3 - 1, dz = k % 3 - 1;
    int x = sC[r][0] + 1 + dx;
    int y = sC[r][1] + 1 + dy;
    int z = sC[r][2] + 1 + dz;
    int key = ((sC[r][3] * HASHD + x) * HASHD + y) * HASHD + z;
    int slot = hash_slot(key);
    int n = -1;
    while (true) {
      int kk = hKey[slot];
      if (kk == key) { n = hVal[slot]; break; }
      if (kk == -1) break;
      slot = (slot + 1) & HMASK;
    }
    sNbr[k][r] = n;
    if (n >= 0) atomicOr(&sHas[k], 1);
  }
  __syncthreads();

  // ---- compact active tap list (uniform) ----
  if (tid == 0) {
    int n = 0;
    for (int k = 0; k < 27; ++k) if (sHas[k]) sAct[n++] = k;
    sAct[27] = n;
  }

  // ---- Stage C: semantic head (VALU) + mask ----
  for (int t = tid; t < 16 * NCLS; t += 256) {
    int r = t / NCLS, j = t % NCLS;
    float s = sem_b[j];
    for (int c = 0; c < C; ++c) s += sF[r * C + c] * sem_w[c * NCLS + j];
    sMask[r][j] = (s > SEM_LOGIT) ? 1.f : 0.f;
    if (r0 + r < N_VOX) out[(size_t)(r0 + r) * NOUT + j] = s;
  }
  __syncthreads();   // publish sAct (and sMask ordering handled by later syncs)

  // ---- Stage D: sparse conv, software-pipelined (ping-pong LDS + prefetch) ----
  v8f acc = {};
  const int rowA = lane & 15;
  const int kbA  = (lane < 16) ? 0 : 8;
  const int nAct = sAct[27];

  gather_tap(feats, sNbr, sAct[0], tid, sG0);   // prologue (center tap always active)
  __syncthreads();
  int p = 0;
  for (int i = 0; i < nAct; ++i) {
    const int k = sAct[i];
    __bf16* cur = p ? sG1 : sG0;
    __bf16* nxt = p ? sG0 : sG1;
    // issue next tap's gather into the other buffer (overlaps with WMMA below)
    if (i + 1 < nAct) gather_tap(feats, sNbr, sAct[i + 1], tid, nxt);
    // prefetch tap i+2's rows into cache (global_prefetch_b8)
    if (i + 2 < nAct) {
      int idx = tid * 8;
      int n2 = sNbr[sAct[i + 2]][idx >> 7];
      if (n2 >= 0) __builtin_prefetch(&feats[(size_t)n2 * C + (idx & 127)], 0, 3);
    }
    #pragma unroll
    for (int kc = 0; kc < 4; ++kc) {
      v16bf a = load_a_frag(&cur[rowA * C + kc * 32 + kbA]);
      v16bf b = load_b_frag(&Wsw[(size_t)((((k * 4 + kc) * 8 + wave) * 32 + lane) << 4)]);
      acc = __builtin_amdgcn_wmma_f32_16x16x32_bf16(false, a, false, b, (short)0, acc, false, false);
    }
    __syncthreads();   // next-buffer writes visible AND current-buffer reads retired
    p ^= 1;
  }

  // ---- Stage E: BN + ELU -> off_feat (f32, LDS) ----
  {
    const int colC = wave * 16 + (lane & 15);
    const int mB   = (lane < 16) ? 0 : 8;
    float g = fo_g[colC], bb = fo_b[colC];
    #pragma unroll
    for (int r = 0; r < 8; ++r) {
      float v = acc[r] * g + bb;
      v = (v > 0.f) ? v : (__expf(v) - 1.f);
      sOF[(mB + r) * C + colC] = v;
    }
  }
  __syncthreads();

  // ---- Stage F: offset MLP (two bf16 WMMA GEMMs) ----
  const __bf16* W1 = Wsw + 27 * 16384;
  const __bf16* W2 = Wsw + 28 * 16384;
  v8f acc1 = {};
  #pragma unroll
  for (int kc = 0; kc < 4; ++kc) {
    v16bf a = load_a_frag(&sA[rowA * C + kc * 32 + kbA]);
    v16bf b = load_b_frag(&W1[(size_t)(((kc * 8 + wave) * 32 + lane) << 4)]);
    acc1 = __builtin_amdgcn_wmma_f32_16x16x32_bf16(false, a, false, b, (short)0, acc1, false, false);
  }
  {
    const int colC = wave * 16 + (lane & 15);
    const int mB   = (lane < 16) ? 0 : 8;
    float g = off_g1[colC], bb = off_b1[colC];
    #pragma unroll
    for (int r = 0; r < 8; ++r) {
      float v = acc1[r] * g + bb;
      v = (v > 0.f) ? v : (__expf(v) - 1.f);
      sH[(mB + r) * C + colC] = (__bf16)v;
    }
  }
  __syncthreads();
  v8f acc2 = {};
  #pragma unroll
  for (int kc = 0; kc < 4; ++kc) {
    v16bf a = load_a_frag(&sH[rowA * C + kc * 32 + kbA]);
    v16bf b = load_b_frag(&W2[(size_t)(((kc * 8 + wave) * 32 + lane) << 4)]);
    acc2 = __builtin_amdgcn_wmma_f32_16x16x32_bf16(false, a, false, b, (short)0, acc2, false, false);
  }
  {
    const int colC = wave * 16 + (lane & 15);
    const int mB   = (lane < 16) ? 0 : 8;
    float g = off_g2[colC], bb = off_b2[colC];
    #pragma unroll
    for (int r = 0; r < 8; ++r) {
      float v = acc2[r] * g + bb;
      v = (v > 0.f) ? v : (__expf(v) - 1.f);
      sF[(mB + r) * C + colC] = v;   // reuse sF as h2 (f32)
    }
  }
  __syncthreads();

  // ---- Stage G: vox_off + voted (clamped) ----
  for (int t = tid; t < 16 * 3; t += 256) {
    int r = t / 3, d = t % 3;
    float s = 0.f;
    for (int c = 0; c < C; ++c) s += sF[r * C + c] * off_w3[c * 3 + d];
    int row = r0 + r;
    if (row < N_VOX) {
      out[(size_t)row * NOUT + 18 + d] = s;
      float mn = ((float)(mm[d]     - 1)) * VSZ;
      float mx = ((float)(mm[3 + d] + 1)) * VSZ;
      float v = (float)sC[r][d] * VSZ + s;
      v = fminf(fmaxf(v, mn), mx);
      out[(size_t)row * NOUT + 21 + d] = v;
    }
  }

  // ---- Stage H: heads (cen / cls / reg) from off_feat ----
  for (int t = tid; t < 16 * 25; t += 256) {
    int r = t / 25, j = t % 25;
    float s = 0.f;
    if (j == 0) {
      for (int c = 0; c < C; ++c) s += sOF[r * C + c] * cen_w[c];
      if (r0 + r < N_VOX) out[(size_t)(r0 + r) * NOUT + 24] = s;
    } else if (j <= NCLS) {
      int cl = j - 1;
      for (int c = 0; c < C; ++c) s += sOF[r * C + c] * cls_w[c * NCLS + cl];
      s = (s + cls_b[cl]) * sMask[r][cl];
      if (r0 + r < N_VOX) out[(size_t)(r0 + r) * NOUT + 25 + cl] = s;
    } else {
      int d = j - 19;
      for (int c = 0; c < C; ++c) s += sOF[r * C + c] * reg_w[c * NREG + d];
      sReg[r][d] = s;
    }
  }
  __syncthreads();

  // ---- Stage I: per-class reg expansion (scales * mask) ----
  for (int t = tid; t < 16 * 108; t += 256) {
    int r = t / 108, q = t % 108;
    int cl = q / 6, d = q % 6;
    int row = r0 + r;
    if (row < N_VOX)
      out[(size_t)row * NOUT + 43 + q] = sReg[r][d] * scales[cl] * sMask[r][cl];
  }
}

extern "C" void kernel_launch(void* const* d_in, const int* in_sizes, int n_in,
                              void* d_out, int out_size, void* d_ws, size_t ws_size,
                              hipStream_t stream) {
  const float* feats  = (const float*)d_in[0];
  const int*   coords = (const int*)d_in[1];
  const int*   bidx   = (const int*)d_in[2];
  const float* off_w1 = (const float*)d_in[3];
  const float* off_g1 = (const float*)d_in[4];
  const float* off_b1 = (const float*)d_in[5];
  const float* off_w2 = (const float*)d_in[6];
  const float* off_g2 = (const float*)d_in[7];
  const float* off_b2 = (const float*)d_in[8];
  const float* off_w3 = (const float*)d_in[9];
  const float* fo_w   = (const float*)d_in[10];
  const float* fo_g   = (const float*)d_in[11];
  const float* fo_b   = (const float*)d_in[12];
  const float* sem_w  = (const float*)d_in[13];
  const float* sem_b  = (const float*)d_in[14];
  const float* cen_w  = (const float*)d_in[15];
  const float* cls_w  = (const float*)d_in[16];
  const float* cls_b  = (const float*)d_in[17];
  const float* reg_w  = (const float*)d_in[18];
  const float* scales = (const float*)d_in[19];
  float* out = (float*)d_out;

  // workspace layout: hKey[HS] | hVal[HS] | mm[6] (+pad) | bf16 swizzled weights (29*16384)
  char* ws = (char*)d_ws;
  int* hKey = (int*)ws;
  int* hVal = hKey + HS;
  int* mm   = (int*)(ws + (size_t)2 * HS * 4);
  __bf16* Wsw = (__bf16*)(ws + (size_t)2 * HS * 4 + 64);

  init_kernel<<<(HS + 255) / 256, 256, 0, stream>>>(hKey, hVal, mm);
  swizzle_kernel<<<(29 * 16384 + 255) / 256, 256, 0, stream>>>(fo_w, off_w1, off_w2, Wsw);
  build_kernel<<<(N_VOX + 255) / 256, 256, 0, stream>>>(coords, bidx, hKey, hVal, mm);
  fused_kernel<<<(N_VOX + 15) / 16, 256, 0, stream>>>(
      feats, coords, bidx,
      off_g1, off_b1, off_g2, off_b2, off_w3,
      fo_g, fo_b, sem_w, sem_b, cen_w, cls_w, cls_b, reg_w, scales,
      hKey, hVal, mm, Wsw, out);
}
